// Navigator_81037442940976
// MI455X (gfx1250) — matrix operand order repaired and checked by reading, not compiled, not run on hardware
//
#include <hip/hip_runtime.h>
#include <hip/hip_bf16.h>

typedef __attribute__((ext_vector_type(16))) _Float16 v16h;
typedef __attribute__((ext_vector_type(8)))  _Float16 v8h;
typedef __attribute__((ext_vector_type(8)))  float    v8f;

namespace {

constexpr int Bc = 2, Tc = 12, Nc = 20000, Fc = 9, Hc = 64, EVc = 9, Ec = 200000;
// packed-weight fragment counts per timestep: W1l(4) W1r(4) W2l(8) W2r(8) W3l(2) W3r(2)
constexpr int FRAGS_PER_T = 28;   // each fragment = 32 lanes * 16 halves = 512 f16

__global__ void k_zero(float* __restrict__ p, long n) {
  long i = (long)blockIdx.x * blockDim.x + threadIdx.x;
  if (i < n) p[i] = 0.f;
}

__global__ void k_deg_count(const int* __restrict__ dst, float* __restrict__ deg, int e) {
  int i = blockIdx.x * blockDim.x + threadIdx.x;
  if (i < e) atomicAdd(&deg[dst[i]], 1.0f);
}

__global__ void k_deg_inv(float* __restrict__ deg, int n) {
  int i = blockIdx.x * blockDim.x + threadIdx.x;
  if (i < n) deg[i] = 1.0f / fmaxf(deg[i], 1.0f);
}

// Fold (x @ encW + encb) @ decW + decb into: dot(h, m1) + s_b + c (exact linear algebra).
// fuse[0..8]=m1, fuse[9]=encb@decW+decb, fuse[10+b]=target_event[b]@m2
__global__ void k_fuse(const float* __restrict__ encW, const float* __restrict__ encb,
                       const float* __restrict__ decW, const float* __restrict__ decb,
                       const float* __restrict__ tgt, float* __restrict__ fuse) {
  __shared__ float m2[EVc];
  int i = threadIdx.x;
  if (i < Fc) {
    float s = 0.f;
    for (int k = 0; k < 64; ++k) s += encW[i * 64 + k] * decW[k];
    fuse[i] = s;
  } else if (i < Fc + EVc) {
    float s = 0.f;
    for (int k = 0; k < 64; ++k) s += encW[i * 64 + k] * decW[k];
    m2[i - Fc] = s;
  } else if (i == Fc + EVc) {
    float s = decb[0];
    for (int k = 0; k < 64; ++k) s += encb[k] * decW[k];
    fuse[Fc] = s;
  }
  __syncthreads();
  if (i < Bc) {
    float s = 0.f;
    for (int j = 0; j < EVc; ++j) s += tgt[i * EVc + j] * m2[j];
    fuse[10 + i] = s;
  }
}

// Pre-pack every weight matrix of every timestep into exact per-lane WMMA B-fragments
// (f16, zero-padded for K>=KIN and col>=nout). Fragment element e of lane L holds
// W[k0 + 8*(L>=16) + (e&7) + 16*(e>>3), tile*16 + (L&15)]  -- the ISA 7.12.2 32x16 f16 B layout.
__global__ void k_pack_w(const float* __restrict__ W1l, const float* __restrict__ W1r,
                         const float* __restrict__ W2l, const float* __restrict__ W2r,
                         const float* __restrict__ W3l, const float* __restrict__ W3r,
                         _Float16* __restrict__ pack) {
  const long TOT = (long)Tc * FRAGS_PER_T * 512;
  long i = (long)blockIdx.x * blockDim.x + threadIdx.x;
  if (i >= TOT) return;
  int e = (int)(i & 15);
  int lane = (int)((i >> 4) & 31);
  int frag = (int)((i >> 9) % FRAGS_PER_T);
  int t = (int)(i / ((long)FRAGS_PER_T * 512));
  const float* W;
  int KINact, nout, tile, chunk;
  if (frag < 8) {                       // SAGE1: 9 -> 64, 1 chunk, 4 tiles
    int g = frag;
    W = (g >= 4 ? W1r : W1l) + t * Fc * Hc;
    KINact = Fc; nout = Hc; tile = g & 3; chunk = 0;
  } else if (frag < 24) {               // SAGE2: 64 -> 64, 2 chunks, 4 tiles
    int g = frag - 8;
    W = (g >= 8 ? W2r : W2l) + t * Hc * Hc;
    g &= 7;
    KINact = Hc; nout = Hc; tile = g >> 1; chunk = g & 1;
  } else {                              // SAGE3: 64 -> 9, 2 chunks, 1 tile
    int g = frag - 24;
    W = (g >= 2 ? W3r : W3l) + t * Hc * Fc;
    KINact = Hc; nout = Fc; tile = 0; chunk = g & 1;
  }
  int kk = chunk * 32 + (lane >> 4) * 8 + (e & 7) + (e >> 3) * 16;
  int col = tile * 16 + (lane & 15);
  float v = (kk < KINact && col < nout) ? W[kk * nout + col] : 0.f;
  pack[i] = (_Float16)v;
}

// graph[:,t] (f32, stride 9) -> zero-padded f16 [B,N,16]
__global__ void k_pad_graph(const float* __restrict__ gt, long gB,
                            _Float16* __restrict__ xg, long total) {
  long i = (long)blockIdx.x * blockDim.x + threadIdx.x;
  if (i >= total) return;
  int f = (int)(i & 15);
  long nn = i >> 4;
  int n = (int)(nn % Nc);
  int b = (int)(nn / Nc);
  float v = (f < Fc) ? gt[(long)b * gB + (long)n * Fc + f] : 0.f;
  xg[i] = (_Float16)v;
}

// aggh = f16(agg * (1/deg)), done once (instead of per column-tile inside the GEMM)
__global__ void k_cvt_agg(const float* __restrict__ agg, const float* __restrict__ deginv,
                          _Float16* __restrict__ aggh, int S, long total) {
  long i = (long)blockIdx.x * blockDim.x + threadIdx.x;
  if (i >= total) return;
  int n = (int)((i / S) % Nc);
  aggh[i] = (_Float16)(agg[i] * deginv[n]);
}

// agg[b, dst, f] += x[b, src, f] * w[e]   (x is f16; accumulation stays f32 atomics)
__global__ void k_scatter(const _Float16* __restrict__ x, long xB, int xS,
                          const float* __restrict__ w,
                          const int* __restrict__ src, const int* __restrict__ dst,
                          float* __restrict__ agg, long aggB, int aggS,
                          int D, long total) {
  long i = (long)blockIdx.x * blockDim.x + threadIdx.x;
  if (i >= total) return;
  int f = (int)(i % D);
  long rest = i / D;
  int e = (int)(rest % Ec);
  int b = (int)(rest / Ec);
  float v = (float)x[(long)b * xB + (long)src[e] * xS + f] * w[e];
  atomicAdd(&agg[(long)b * aggB + (long)dst[e] * aggS + f], v);
}

// out = ELU( (agg/deg) @ Wl + bias + x @ Wr ) with one 16x16 tile per wave.
// A inputs are pre-padded f16 (unconditional 16B loads); B comes from pre-packed
// fragments (two unconditional 16B loads). No lane-divergent guards in the K loop.
template <int KP, int NOUTPAD, typename OutT>
__global__ __launch_bounds__(128) void k_sage_mm(
    const _Float16* __restrict__ aggh, long aggB, int aggS,
    const _Float16* __restrict__ xh, long xB, int xS,
    const _Float16* __restrict__ Wlp, const _Float16* __restrict__ Wrp,
    const float* __restrict__ bias, int nout,
    OutT* __restrict__ outp, long outB, int outS, int nrows) {
  constexpr int CT = NOUTPAD / 16;   // column tiles covered by the 4 waves
  constexpr int RT = 4 / CT;         // row tiles per block
  constexpr int CH = (KP + 31) / 32; // k chunks
  const int wave = threadIdx.x >> 5;
  const int lane = threadIdx.x & 31;
  const int rowTile = blockIdx.x * RT + (CT == 4 ? 0 : wave);
  const int c0 = (CT == 4) ? wave * 16 : 0;
  const int tileIdx = (CT == 4) ? wave : 0;
  const int r0 = rowTile * 16;
  if (r0 >= nrows) return;  // wave-uniform; EXEC all-ones for WMMA
  const int b = blockIdx.y;
  const int lrow = lane & 15;
  const int halfsel = lane >> 4;
  const int mrow = r0 + lrow;
  const _Float16* aggRow = aggh + (long)b * aggB + (long)mrow * aggS;
  const _Float16* xRow = xh + (long)b * xB + (long)mrow * xS;
  const int col = c0 + lrow;
  __builtin_prefetch(aggRow, 0, 1);
  __builtin_prefetch(xRow, 0, 1);

  v8f acc = {};
#pragma unroll
  for (int ch = 0; ch < CH; ++ch) {
    const int k0 = ch * 32;
    const v8h zero = {};
    // A fragments: elements 0..7 = K k0+8*halfsel+{0..7}; 8..15 = +16 (ISA 7.12.2)
    v8h aAlo = *(const v8h*)(aggRow + k0 + halfsel * 8);
    v8h aXlo = *(const v8h*)(xRow + k0 + halfsel * 8);
    v8h aAhi = (KP >= 32) ? *(const v8h*)(aggRow + k0 + 16 + halfsel * 8) : zero;
    v8h aXhi = (KP >= 32) ? *(const v8h*)(xRow + k0 + 16 + halfsel * 8) : zero;
    v16h aA = __builtin_shufflevector(aAlo, aAhi, 0, 1, 2, 3, 4, 5, 6, 7, 8, 9, 10, 11, 12, 13, 14, 15);
    v16h aX = __builtin_shufflevector(aXlo, aXhi, 0, 1, 2, 3, 4, 5, 6, 7, 8, 9, 10, 11, 12, 13, 14, 15);
    // B fragments: pre-packed, contiguous 32B per lane
    const _Float16* pL = Wlp + ((long)(tileIdx * CH + ch)) * 512 + lane * 16;
    const _Float16* pR = Wrp + ((long)(tileIdx * CH + ch)) * 512 + lane * 16;
    v8h l0 = *(const v8h*)pL, l1 = *(const v8h*)(pL + 8);
    v8h r0v = *(const v8h*)pR, r1 = *(const v8h*)(pR + 8);
    v16h bL = __builtin_shufflevector(l0, l1, 0, 1, 2, 3, 4, 5, 6, 7, 8, 9, 10, 11, 12, 13, 14, 15);
    v16h bR = __builtin_shufflevector(r0v, r1, 0, 1, 2, 3, 4, 5, 6, 7, 8, 9, 10, 11, 12, 13, 14, 15);
    acc = __builtin_amdgcn_wmma_f32_16x16x32_f16(false, aA, false, bL, (short)0, acc, false, false);
    acc = __builtin_amdgcn_wmma_f32_16x16x32_f16(false, aX, false, bR, (short)0, acc, false, false);
  }

  const float bv = (col < nout) ? bias[col] : 0.f;
  OutT* outBase = outp + (long)b * outB;
#pragma unroll
  for (int j = 0; j < 8; ++j) {
    const int rr = r0 + j + halfsel * 8;
    float v = acc[j] + bv;
    v = (v > 0.f) ? v : expm1f(v);  // ELU, alpha=1
    if (col < nout) outBase[(long)rr * outS + col] = (OutT)v;
  }
}

// GRU cell (9->9) + fused linear head:  out[b,t,n] = (h_new . m1 + s_b + c) * mask
__global__ void k_gru(const float* __restrict__ x9,
                      const float* __restrict__ graph_t, long gB,
                      const float* __restrict__ Wih, const float* __restrict__ Whh,
                      const float* __restrict__ bih, const float* __restrict__ bhh,
                      const float* __restrict__ fuse,
                      float* __restrict__ h, float* __restrict__ outp, int t) {
  long i = (long)blockIdx.x * blockDim.x + threadIdx.x;
  if (i >= (long)Bc * Nc) return;
  int b = (int)(i / Nc), n = (int)(i % Nc);
  const float* xp = x9 + ((long)b * Nc + n) * Fc;
  float* hp = h + ((long)b * Nc + n) * Fc;
  float xv[Fc], hv[Fc];
#pragma unroll
  for (int q = 0; q < Fc; ++q) { xv[q] = xp[q]; hv[q] = hp[q]; }
  float gi[3 * Fc], gh[3 * Fc];
#pragma unroll
  for (int j = 0; j < 3 * Fc; ++j) { gi[j] = bih[j]; gh[j] = bhh[j]; }
  for (int k = 0; k < Fc; ++k) {
    const float xk = xv[k], hk = hv[k];
#pragma unroll
    for (int j = 0; j < 3 * Fc; ++j) {
      gi[j] += xk * Wih[k * 3 * Fc + j];
      gh[j] += hk * Whh[k * 3 * Fc + j];
    }
  }
  float acc = 0.f;
#pragma unroll
  for (int q = 0; q < Fc; ++q) {
    float r = 1.f / (1.f + expf(-(gi[q] + gh[q])));
    float z = 1.f / (1.f + expf(-(gi[Fc + q] + gh[Fc + q])));
    float nn = tanhf(gi[2 * Fc + q] + r * gh[2 * Fc + q]);
    float hq = (1.f - z) * nn + z * hv[q];
    hp[q] = hq;
    acc += hq * fuse[q];
  }
  float mask = (graph_t[(long)b * gB + (long)n * Fc] != 0.f) ? 1.f : 0.f;
  outp[((long)b * Tc + t) * (long)Nc + n] = (acc + fuse[10 + b] + fuse[Fc]) * mask;
}

}  // namespace

extern "C" void kernel_launch(void* const* d_in, const int* in_sizes, int n_in,
                              void* d_out, int out_size, void* d_ws, size_t ws_size,
                              hipStream_t stream) {
  (void)in_sizes; (void)n_in; (void)out_size; (void)ws_size;
  const float* graph = (const float*)d_in[0];
  const float* tgt   = (const float*)d_in[1];
  // d_in[2] input_event: unused by the reference forward pass
  const float* ew    = (const float*)d_in[3];
  const int*   esrc  = (const int*)d_in[4];
  const int*   edst  = (const int*)d_in[5];
  const float* W1l = (const float*)d_in[6];
  const float* b1  = (const float*)d_in[7];
  const float* W1r = (const float*)d_in[8];
  const float* W2l = (const float*)d_in[9];
  const float* b2  = (const float*)d_in[10];
  const float* W2r = (const float*)d_in[11];
  const float* W3l = (const float*)d_in[12];
  const float* b3  = (const float*)d_in[13];
  const float* W3r = (const float*)d_in[14];
  const float* Wih = (const float*)d_in[15];
  const float* Whh = (const float*)d_in[16];
  const float* bih = (const float*)d_in[17];
  const float* bhh = (const float*)d_in[18];
  const float* encW = (const float*)d_in[19];
  const float* encb = (const float*)d_in[20];
  const float* decW = (const float*)d_in[21];
  const float* decb = (const float*)d_in[22];
  float* outp = (float*)d_out;

  // Workspace layout (all offsets multiples of 16 floats -> 64B aligned; ~30 MB, L2-resident).
  float* ws       = (float*)d_ws;
  float* deginv   = ws;                                       // N (20000)
  float* fuse     = deginv + Nc;                              // 16
  _Float16* packW = (_Float16*)(fuse + 16);                   // T*28*512 f16 = 86016 fl
  float* agg      = (float*)(packW + (long)Tc * FRAGS_PER_T * 512);  // B*N*64 f32
  _Float16* xg    = (_Float16*)(agg + (long)Bc * Nc * Hc);    // B*N*16 f16
  _Float16* x1h   = xg + (long)Bc * Nc * 16;                  // B*N*64 f16
  _Float16* x2h   = x1h + (long)Bc * Nc * Hc;                 // B*N*64 f16
  _Float16* aggh  = x2h + (long)Bc * Nc * Hc;                 // B*N*64 f16
  float* x9       = (float*)(aggh + (long)Bc * Nc * Hc);      // B*N*9 f32
  float* hb       = x9 + (long)Bc * Nc * Fc;                  // B*N*9 f32

  const int TB = 256;
  auto blocks = [](long n, int tb) { return dim3((unsigned)((n + tb - 1) / tb)); };

  // One-time per call: 1/deg, h0 = 0, fused head coefficients, packed f16 weight fragments.
  k_zero<<<blocks(Nc, TB), TB, 0, stream>>>(deginv, Nc);
  k_deg_count<<<blocks(Ec, TB), TB, 0, stream>>>(edst, deginv, Ec);
  k_deg_inv<<<blocks(Nc, TB), TB, 0, stream>>>(deginv, Nc);
  k_zero<<<blocks((long)Bc * Nc * Fc, TB), TB, 0, stream>>>(hb, (long)Bc * Nc * Fc);
  k_fuse<<<1, 32, 0, stream>>>(encW, encb, decW, decb, tgt, fuse);
  k_pack_w<<<blocks((long)Tc * FRAGS_PER_T * 512, TB), TB, 0, stream>>>(
      W1l, W1r, W2l, W2r, W3l, W3r, packW);

  const long gB = (long)Tc * Nc * Fc;
  const long fragT = (long)FRAGS_PER_T * 512;
  for (int t = 0; t < Tc; ++t) {
    const float* gt = graph + (long)t * Nc * Fc;
    const _Float16* pT = packW + (long)t * fragT;

    // SAGE1: 9 -> 64 (K padded to 16)
    k_pad_graph<<<blocks((long)Bc * Nc * 16, TB), TB, 0, stream>>>(gt, gB, xg, (long)Bc * Nc * 16);
    k_zero<<<blocks((long)Bc * Nc * 16, TB), TB, 0, stream>>>(agg, (long)Bc * Nc * 16);
    k_scatter<<<blocks((long)Bc * Ec * Fc, TB), TB, 0, stream>>>(
        xg, (long)Nc * 16, 16, ew, esrc, edst, agg, (long)Nc * 16, 16, Fc, (long)Bc * Ec * Fc);
    k_cvt_agg<<<blocks((long)Bc * Nc * 16, TB), TB, 0, stream>>>(
        agg, deginv, aggh, 16, (long)Bc * Nc * 16);
    k_sage_mm<16, 64, _Float16><<<dim3(Nc / 16, Bc), 128, 0, stream>>>(
        aggh, (long)Nc * 16, 16, xg, (long)Nc * 16, 16,
        pT + 0 * 512, pT + 4 * 512, b1 + t * Hc, Hc,
        x1h, (long)Nc * Hc, Hc, Nc);

    // SAGE2: 64 -> 64
    k_zero<<<blocks((long)Bc * Nc * Hc, TB), TB, 0, stream>>>(agg, (long)Bc * Nc * Hc);
    k_scatter<<<blocks((long)Bc * Ec * Hc, TB), TB, 0, stream>>>(
        x1h, (long)Nc * Hc, Hc, ew, esrc, edst, agg, (long)Nc * Hc, Hc, Hc, (long)Bc * Ec * Hc);
    k_cvt_agg<<<blocks((long)Bc * Nc * Hc, TB), TB, 0, stream>>>(
        agg, deginv, aggh, Hc, (long)Bc * Nc * Hc);
    k_sage_mm<64, 64, _Float16><<<dim3(Nc / 16, Bc), 128, 0, stream>>>(
        aggh, (long)Nc * Hc, Hc, x1h, (long)Nc * Hc, Hc,
        pT + 8 * 512, pT + 16 * 512, b2 + t * Hc, Hc,
        x2h, (long)Nc * Hc, Hc, Nc);

    // SAGE3: 64 -> 9 (one 16-wide column tile, cols >= 9 masked at store only)
    k_zero<<<blocks((long)Bc * Nc * Hc, TB), TB, 0, stream>>>(agg, (long)Bc * Nc * Hc);
    k_scatter<<<blocks((long)Bc * Ec * Hc, TB), TB, 0, stream>>>(
        x2h, (long)Nc * Hc, Hc, ew, esrc, edst, agg, (long)Nc * Hc, Hc, Hc, (long)Bc * Ec * Hc);
    k_cvt_agg<<<blocks((long)Bc * Nc * Hc, TB), TB, 0, stream>>>(
        agg, deginv, aggh, Hc, (long)Bc * Nc * Hc);
    k_sage_mm<64, 16, float><<<dim3((Nc / 16 + 3) / 4, Bc), 128, 0, stream>>>(
        aggh, (long)Nc * Hc, Hc, x2h, (long)Nc * Hc, Hc,
        pT + 24 * 512, pT + 26 * 512, b3 + t * Fc, Fc,
        x9, (long)Nc * Fc, Fc, Nc);

    // GRU + fused encoder/decoder epilogue -> out[:, t]
    k_gru<<<blocks((long)Bc * Nc, TB), TB, 0, stream>>>(
        x9, gt, gB, Wih + t * Fc * 3 * Fc, Whh + t * Fc * 3 * Fc,
        bih + t * 3 * Fc, bhh + t * 3 * Fc, fuse, hb, outp, t);
  }
}
